// FakeExperts_62878321214300
// MI455X (gfx1250) — matrix operand order repaired and checked by reading, not compiled, not run on hardware
//
#include <hip/hip_runtime.h>

namespace {

constexpr int E = 8;
constexpr int T = 4096;
constexpr int H = 2048;
constexpr int I = 2048;
constexpr float kAlpha = 1.702f;
constexpr float kLimit = 7.0f;
constexpr int KC = 32;  // K-chunk per LDS stage (one WMMA K)

typedef __attribute__((ext_vector_type(16))) __bf16 v16bf;
typedef __attribute__((ext_vector_type(8)))  float  v8f;

union Frag {
  v16bf bf;
  uint4 q[2];
};

// Round-to-nearest-even f32 -> bf16 bits.
__device__ __forceinline__ unsigned int bf16_rne_bits(float x) {
  unsigned int u = __float_as_uint(x);
  return (u + 0x7FFFu + ((u >> 16) & 1u)) >> 16;
}

// Split two consecutive-K f32 values into a packed-bf16 "hi" word and a
// packed-bf16 residual "lo" word (K-even in bits [15:0], K-odd in [31:16]),
// matching the 16-bit WMMA operand packing.
__device__ __forceinline__ void split2(float a, float b,
                                       unsigned int& hi, unsigned int& lo) {
  unsigned int ha = bf16_rne_bits(a);
  unsigned int hb = bf16_rne_bits(b);
  float fa = __uint_as_float(ha << 16);
  float fb = __uint_as_float(hb << 16);
  hi = ha | (hb << 16);
  lo = bf16_rne_bits(a - fa) | (bf16_rne_bits(b - fb) << 16);
}

__device__ __forceinline__ v8f wmma_bf16(const Frag& a, const Frag& b, v8f c) {
  return __builtin_amdgcn_wmma_f32_16x16x32_bf16(
      false, a.bf, false, b.bf, (short)0, c, false, false);
}

// A-matrix 16x32 bf16 fragment (ISA 7.12.2): lane group g (lane>>4):
//   VGPR 0..3 hold K-pairs (v + 4g), VGPR 4..7 hold K-pairs (8 + (v-4) + 4g).
// LDS row = 16 packed-K-pair dwords, so fragment = two ds_read_b128.
__device__ __forceinline__ void load_fragA(Frag& f, const unsigned int* row16,
                                           int g) {
  f.q[0] = *(const uint4*)(row16 + 4 * g);
  f.q[1] = *(const uint4*)(row16 + 8 + 4 * g);
}

// B-matrix 32x16 bf16 fragment: lane n = lane&15, VGPR 0..3 hold K-pairs (v + 4g),
// VGPR 4..7 hold K-pairs (8 + (v-4) + 4g).
// LDS stored [n][kp] -> fragment = two ds_read_b128.
__device__ __forceinline__ void load_fragB(Frag& f, const unsigned int* row16,
                                           int g) {
  f.q[0] = *(const uint4*)(row16 + 4 * g);
  f.q[1] = *(const uint4*)(row16 + 8 + 4 * g);
}

// ---------------------------------------------------------------------------
// GEMM1: gate_up = x @ W1 (+bias), fused clamped-GLU -> h (f32) in workspace.
// WG tile: 128 rows x 128 W-columns (= 64 h-columns, gate/up de-interleaved
// into LDS-n [0,64) = gate, [64,128) = up).
// ---------------------------------------------------------------------------
__global__ void __launch_bounds__(256)
gate_up_glu_kernel(const float* __restrict__ x,
                   const float* __restrict__ wgu,
                   const float* __restrict__ bgu,
                   float* __restrict__ hbuf) {
  __shared__ unsigned int Ahi[2][128][16];
  __shared__ unsigned int Alo[2][128][16];
  __shared__ unsigned int Bhi[2][128][16];
  __shared__ unsigned int Blo[2][128][16];

  const int e  = blockIdx.z;
  const int m0 = blockIdx.y * 128;
  const int j0 = blockIdx.x * 64;  // h-column base

  const int t    = threadIdx.x;
  const int wave = t >> 5;
  const int lane = t & 31;
  const int g    = lane >> 4;
  const int ml   = lane & 15;
  const int wm   = wave & 3;   // 4 row groups of 32
  const int wn   = wave >> 2;  // 2 h-col groups of 32

  const float* xe = x + (size_t)e * T * H;
  const float* we = wgu + (size_t)e * H * (2 * I);

  // Staging coordinates.
  const int ar   = t >> 1;   // A row 0..127
  const int ah   = t & 1;    // K half (16 floats)
  const int bn   = t & 127;  // LDS B row (de-interleaved)
  const int bh   = t >> 7;   // K half
  const int bcol = 2 * (j0 + (bn & 63)) + (bn >> 6);  // global W column

  v8f acc[2][2][2];  // [mi][sel: 0=gate,1=up][ci]
  v8f zero = {0.f, 0.f, 0.f, 0.f, 0.f, 0.f, 0.f, 0.f};
#pragma unroll
  for (int mi = 0; mi < 2; ++mi)
#pragma unroll
    for (int s = 0; s < 2; ++s)
#pragma unroll
      for (int ci = 0; ci < 2; ++ci) acc[mi][s][ci] = zero;

  const int nk = H / KC;

  auto stage = [&](int kt, int buf) {
    const int k0 = kt * KC;
    {  // A tile: 128 x 32 f32 -> split bf16 hi/lo, packed K-pairs.
      const float* ap = xe + (size_t)(m0 + ar) * H + k0 + 16 * ah;
      __builtin_prefetch(ap + KC, 0, 0);
      float4 v0 = *(const float4*)(ap + 0);
      float4 v1 = *(const float4*)(ap + 4);
      float4 v2 = *(const float4*)(ap + 8);
      float4 v3 = *(const float4*)(ap + 12);
      const int kp = 8 * ah;
      split2(v0.x, v0.y, Ahi[buf][ar][kp + 0], Alo[buf][ar][kp + 0]);
      split2(v0.z, v0.w, Ahi[buf][ar][kp + 1], Alo[buf][ar][kp + 1]);
      split2(v1.x, v1.y, Ahi[buf][ar][kp + 2], Alo[buf][ar][kp + 2]);
      split2(v1.z, v1.w, Ahi[buf][ar][kp + 3], Alo[buf][ar][kp + 3]);
      split2(v2.x, v2.y, Ahi[buf][ar][kp + 4], Alo[buf][ar][kp + 4]);
      split2(v2.z, v2.w, Ahi[buf][ar][kp + 5], Alo[buf][ar][kp + 5]);
      split2(v3.x, v3.y, Ahi[buf][ar][kp + 6], Alo[buf][ar][kp + 6]);
      split2(v3.z, v3.w, Ahi[buf][ar][kp + 7], Alo[buf][ar][kp + 7]);
    }
    {  // B tile: 32(K) x 128(N, de-interleaved), packed K-pairs per dword.
      const float* bp = we + (size_t)(k0 + 16 * bh) * (2 * I) + bcol;
#pragma unroll
      for (int kk = 0; kk < 8; ++kk) {
        float b0 = bp[0];
        float b1 = bp[2 * I];
        split2(b0, b1, Bhi[buf][bn][8 * bh + kk], Blo[buf][bn][8 * bh + kk]);
        bp += 4 * I;
      }
    }
  };

  stage(0, 0);
  for (int kt = 0; kt < nk; ++kt) {
    __syncthreads();
    const int buf = kt & 1;
    if (kt + 1 < nk) stage(kt + 1, buf ^ 1);

    Frag fah[2], fal[2];
#pragma unroll
    for (int mi = 0; mi < 2; ++mi) {
      const int row = 32 * wm + 16 * mi + ml;
      load_fragA(fah[mi], &Ahi[buf][row][0], g);
      load_fragA(fal[mi], &Alo[buf][row][0], g);
    }
    Frag fbh[2][2], fbl[2][2];
#pragma unroll
    for (int s = 0; s < 2; ++s)
#pragma unroll
      for (int ci = 0; ci < 2; ++ci) {
        const int n = 64 * s + 32 * wn + 16 * ci + ml;
        load_fragB(fbh[s][ci], &Bhi[buf][n][0], g);
        load_fragB(fbl[s][ci], &Blo[buf][n][0], g);
      }
#pragma unroll
    for (int mi = 0; mi < 2; ++mi)
#pragma unroll
      for (int s = 0; s < 2; ++s)
#pragma unroll
        for (int ci = 0; ci < 2; ++ci) {
          v8f a0 = acc[mi][s][ci];
          a0 = wmma_bf16(fah[mi], fbh[s][ci], a0);  // hi*hi
          a0 = wmma_bf16(fah[mi], fbl[s][ci], a0);  // hi*lo
          a0 = wmma_bf16(fal[mi], fbh[s][ci], a0);  // lo*hi
          acc[mi][s][ci] = a0;
        }
  }

  // Fused epilogue: clamp + GLU, write h (f32) to workspace.
  const float* be = bgu + (size_t)e * 2 * I;
#pragma unroll
  for (int mi = 0; mi < 2; ++mi)
#pragma unroll
    for (int ci = 0; ci < 2; ++ci) {
      const int j = j0 + 32 * wn + 16 * ci + ml;
      const float gb = be[2 * j];
      const float ub = be[2 * j + 1];
      const int mbase = m0 + 32 * wm + 16 * mi + 8 * g;
#pragma unroll
      for (int r = 0; r < 8; ++r) {
        float gate = acc[mi][0][ci][r] + gb;
        float up   = acc[mi][1][ci][r] + ub;
        gate = fminf(gate, kLimit);
        up   = fminf(fmaxf(up, -kLimit), kLimit);
        float glu = gate / (1.0f + __expf(-kAlpha * gate));
        float hv  = (up + 1.0f) * glu;
        hbuf[(size_t)(e * T + mbase + r) * I + j] = hv;
      }
    }
}

// ---------------------------------------------------------------------------
// GEMM2: out = h @ W2 + bias.  WG tile 128x128, wave tile 32x64.
// ---------------------------------------------------------------------------
__global__ void __launch_bounds__(256)
down_proj_kernel(const float* __restrict__ hbuf,
                 const float* __restrict__ wd,
                 const float* __restrict__ bd,
                 float* __restrict__ out) {
  __shared__ unsigned int Ahi[2][128][16];
  __shared__ unsigned int Alo[2][128][16];
  __shared__ unsigned int Bhi[2][128][16];
  __shared__ unsigned int Blo[2][128][16];

  const int e  = blockIdx.z;
  const int m0 = blockIdx.y * 128;
  const int n0 = blockIdx.x * 128;

  const int t    = threadIdx.x;
  const int wave = t >> 5;
  const int lane = t & 31;
  const int g    = lane >> 4;
  const int ml   = lane & 15;
  const int wm   = wave & 3;   // 4 row groups of 32
  const int wn   = wave >> 2;  // 2 col groups of 64

  const float* he = hbuf + (size_t)e * T * I;
  const float* we = wd + (size_t)e * I * H;

  const int ar = t >> 1;
  const int ah = t & 1;
  const int bn = t & 127;
  const int bh = t >> 7;

  v8f acc[2][4];
  v8f zero = {0.f, 0.f, 0.f, 0.f, 0.f, 0.f, 0.f, 0.f};
#pragma unroll
  for (int mi = 0; mi < 2; ++mi)
#pragma unroll
    for (int ci = 0; ci < 4; ++ci) acc[mi][ci] = zero;

  const int nk = I / KC;

  auto stage = [&](int kt, int buf) {
    const int k0 = kt * KC;
    {
      const float* ap = he + (size_t)(m0 + ar) * I + k0 + 16 * ah;
      __builtin_prefetch(ap + KC, 0, 0);
      float4 v0 = *(const float4*)(ap + 0);
      float4 v1 = *(const float4*)(ap + 4);
      float4 v2 = *(const float4*)(ap + 8);
      float4 v3 = *(const float4*)(ap + 12);
      const int kp = 8 * ah;
      split2(v0.x, v0.y, Ahi[buf][ar][kp + 0], Alo[buf][ar][kp + 0]);
      split2(v0.z, v0.w, Ahi[buf][ar][kp + 1], Alo[buf][ar][kp + 1]);
      split2(v1.x, v1.y, Ahi[buf][ar][kp + 2], Alo[buf][ar][kp + 2]);
      split2(v1.z, v1.w, Ahi[buf][ar][kp + 3], Alo[buf][ar][kp + 3]);
      split2(v2.x, v2.y, Ahi[buf][ar][kp + 4], Alo[buf][ar][kp + 4]);
      split2(v2.z, v2.w, Ahi[buf][ar][kp + 5], Alo[buf][ar][kp + 5]);
      split2(v3.x, v3.y, Ahi[buf][ar][kp + 6], Alo[buf][ar][kp + 6]);
      split2(v3.z, v3.w, Ahi[buf][ar][kp + 7], Alo[buf][ar][kp + 7]);
    }
    {
      const float* bp = we + (size_t)(k0 + 16 * bh) * H + n0 + bn;
#pragma unroll
      for (int kk = 0; kk < 8; ++kk) {
        float b0 = bp[0];
        float b1 = bp[H];
        split2(b0, b1, Bhi[buf][bn][8 * bh + kk], Blo[buf][bn][8 * bh + kk]);
        bp += 2 * H;
      }
    }
  };

  stage(0, 0);
  for (int kt = 0; kt < nk; ++kt) {
    __syncthreads();
    const int buf = kt & 1;
    if (kt + 1 < nk) stage(kt + 1, buf ^ 1);

    Frag fah[2], fal[2];
#pragma unroll
    for (int mi = 0; mi < 2; ++mi) {
      const int row = 32 * wm + 16 * mi + ml;
      load_fragA(fah[mi], &Ahi[buf][row][0], g);
      load_fragA(fal[mi], &Alo[buf][row][0], g);
    }
    Frag fbh[4], fbl[4];
#pragma unroll
    for (int ci = 0; ci < 4; ++ci) {
      const int n = 64 * wn + 16 * ci + ml;
      load_fragB(fbh[ci], &Bhi[buf][n][0], g);
      load_fragB(fbl[ci], &Blo[buf][n][0], g);
    }
#pragma unroll
    for (int mi = 0; mi < 2; ++mi)
#pragma unroll
      for (int ci = 0; ci < 4; ++ci) {
        v8f a0 = acc[mi][ci];
        a0 = wmma_bf16(fah[mi], fbh[ci], a0);
        a0 = wmma_bf16(fah[mi], fbl[ci], a0);
        a0 = wmma_bf16(fal[mi], fbh[ci], a0);
        acc[mi][ci] = a0;
      }
  }

#pragma unroll
  for (int mi = 0; mi < 2; ++mi)
#pragma unroll
    for (int ci = 0; ci < 4; ++ci) {
      const int nc = n0 + 64 * wn + 16 * ci + ml;
      const float bias = bd[(size_t)e * H + nc];
      const int mbase = m0 + 32 * wm + 16 * mi + 8 * g;
#pragma unroll
      for (int r = 0; r < 8; ++r)
        out[(size_t)(e * T + mbase + r) * H + nc] = acc[mi][ci][r] + bias;
    }
}

}  // namespace

extern "C" void kernel_launch(void* const* d_in, const int* in_sizes, int n_in,
                              void* d_out, int out_size, void* d_ws,
                              size_t ws_size, hipStream_t stream) {
  (void)in_sizes; (void)n_in; (void)out_size; (void)ws_size;
  const float* x   = (const float*)d_in[0];
  const float* wgu = (const float*)d_in[1];
  const float* bgu = (const float*)d_in[2];
  const float* wd  = (const float*)d_in[3];
  const float* bd  = (const float*)d_in[4];
  float* out  = (float*)d_out;
  float* hbuf = (float*)d_ws;  // E*T*I f32 intermediate (256 MB)

  dim3 block(256);
  dim3 g1(I / 64, T / 128, E);
  dim3 g2(H / 128, T / 128, E);
  hipLaunchKernelGGL(gate_up_glu_kernel, g1, block, 0, stream, x, wgu, bgu, hbuf);
  hipLaunchKernelGGL(down_proj_kernel,  g2, block, 0, stream, hbuf, wd, bd, out);
}
